// GMN_84945863180929
// MI455X (gfx1250) — matrix operand order repaired and checked by reading, not compiled, not run on hardware
//
#include <hip/hip_runtime.h>
#include <hip/hip_bf16.h>

// ---------------------------------------------------------------------------
// GMN for MI455X (gfx1250, wave32).
// Dominant op: fused embedding-gather + dual skinny GEMM (K=768, N=10x2)
// using v_wmma_f32_16x16x32_f16 with B-fragments staged in LDS (320KB/WGP).
// Everything else is tiny per-graph work.
// ---------------------------------------------------------------------------

#define VOCAB 21128
#define Dm    768
#define Sm    50
#define Tm    5
#define Bm    512
#define Em    128
#define Cm    10

#define BFRAG_DW (24 * 32 * 8)   // 6144 dwords per weight matrix

typedef __attribute__((ext_vector_type(16))) _Float16     v16h;
typedef __attribute__((ext_vector_type(8)))  float        v8f;
typedef __attribute__((ext_vector_type(4)))  float        f32x4;

__device__ __forceinline__ unsigned int pack_h2(float a, float b) {
    _Float16 ha = (_Float16)a, hb = (_Float16)b;
    unsigned short ua, ub;
    __builtin_memcpy(&ua, &ha, 2);
    __builtin_memcpy(&ub, &hb, 2);
    return (unsigned int)ua | ((unsigned int)ub << 16);
}

// ---------------------------------------------------------------------------
// K0: pack conv_W [768,10] and cross_w [768,10] into per-lane B-fragments
// for v_wmma_f32_16x16x32_f16 (N padded 10->16 with zeros).
// Layout per 32-K chunk c: lane l holds column n = l&15; lanes 0-15 hold
// K = c*32 + 0..15 (2 halves per dword), lanes 16-31 hold K = c*32 + 16..31.
// ---------------------------------------------------------------------------
__global__ void gmn_prep_bfrag(const float* __restrict__ convW,
                               const float* __restrict__ crossW,
                               unsigned int* __restrict__ bfC,
                               unsigned int* __restrict__ bfX) {
    int d = blockIdx.x * blockDim.x + threadIdx.x;
    if (d >= BFRAG_DW) return;
    int c    = d >> 8;          // K-chunk
    int lane = (d >> 3) & 31;
    int j    = d & 7;           // dword within fragment
    int n    = lane & 15;
    int kb   = (lane & 16) ? 16 : 0;
    int k0   = c * 32 + kb + 2 * j;
    float c0 = 0.f, c1 = 0.f, x0 = 0.f, x1 = 0.f;
    if (n < Cm) {
        c0 = convW[k0 * Cm + n];  c1 = convW[(k0 + 1) * Cm + n];
        x0 = crossW[k0 * Cm + n]; x1 = crossW[(k0 + 1) * Cm + n];
    }
    bfC[d] = pack_h2(c0, c1);
    bfX[d] = pack_h2(x0, x1);
}

// ---------------------------------------------------------------------------
// K1: fused gather + dual projection.  One wave per 16-row tile.
// A fragment in contiguous k order (same as B packing):
//   lanes 0-15 : M = lane,    K in {0..15}  of the chunk
//   lanes 16-31: M = lane-16, K in {16..31} of the chunk
// B fragments staged once per block into LDS (48KB); WMMA loop reads them
// via ds_load_b128 while the vmem pipe streams the gathered rows from L2.
// ---------------------------------------------------------------------------
__global__ void gmn_proj(const int* __restrict__ tokens, int n_tiles,
                         const float* __restrict__ emb,
                         const unsigned int* __restrict__ bfC,
                         const unsigned int* __restrict__ bfX,
                         float* __restrict__ outC,
                         float* __restrict__ outX) {
    __shared__ __align__(32) unsigned int sB[2][BFRAG_DW];

    // Cooperative fill: 2 * 6144 dwords by all threads (uint4 vectors).
    {
        const uint4* srcC = (const uint4*)bfC;
        const uint4* srcX = (const uint4*)bfX;
        uint4* dstC = (uint4*)&sB[0][0];
        uint4* dstX = (uint4*)&sB[1][0];
        for (int i = threadIdx.x; i < BFRAG_DW / 4; i += blockDim.x) {
            dstC[i] = srcC[i];
            dstX[i] = srcX[i];
        }
    }
    __syncthreads();

    int wave = threadIdx.x >> 5;
    int lane = threadIdx.x & 31;
    int tile = blockIdx.x * (blockDim.x >> 5) + wave;
    if (tile >= n_tiles) return;   // wave-uniform; no barriers after this point

    int rowbase = tile << 4;
    int m  = lane & 15;
    long tok = (long)tokens[rowbase + m];
    const float* row = emb + tok * Dm;
    int kb = (lane & 16) ? 16 : 0;
    const unsigned int* sBc = &sB[0][lane * 8];
    const unsigned int* sBx = &sB[1][lane * 8];

    v8f accC = {};
    v8f accX = {};

    for (int c = 0; c < 24; ++c) {
        if (c + 1 < 24)
            __builtin_prefetch((const void*)(row + (c + 1) * 32), 0, 0);

        const f32x4* p = (const f32x4*)(row + c * 32 + kb);
        f32x4 f0 = p[0];
        f32x4 f1 = p[1];
        const f32x4* q = (const f32x4*)(row + c * 32 + kb + 8);
        f32x4 f2 = q[0];
        f32x4 f3 = q[1];

        v16h a;
#pragma unroll
        for (int j = 0; j < 4; ++j) {
            a[j]      = (_Float16)f0[j];
            a[4 + j]  = (_Float16)f1[j];
            a[8 + j]  = (_Float16)f2[j];
            a[12 + j] = (_Float16)f3[j];
        }

        v16h bc = *(const v16h*)(sBc + (size_t)c * 256);
        v16h bx = *(const v16h*)(sBx + (size_t)c * 256);

        accC = __builtin_amdgcn_wmma_f32_16x16x32_f16(
            false, a, false, bc, (short)0, accC, false, false);
        accX = __builtin_amdgcn_wmma_f32_16x16x32_f16(
            false, a, false, bx, (short)0, accX, false, false);
    }

    // D layout: VGPR r -> M = r (lanes 0-15, N=lane) / M = 8+r (lanes 16-31)
    int n  = lane & 15;
    int rb = rowbase + ((lane & 16) ? 8 : 0);
    if (n < Cm) {
#pragma unroll
        for (int r = 0; r < 8; ++r) {
            outC[(long)(rb + r) * Cm + n] = accC[r];
            outX[(long)(rb + r) * Cm + n] = accX[r];
        }
    }
}

// ---------------------------------------------------------------------------
// K2: GCN aggregation (self-loops + symmetric norm), deterministic per-dst
// scan over 128 edges. One block per graph (2560 ctx graphs + 512 resp).
// ---------------------------------------------------------------------------
__global__ void gmn_gcn(const int* __restrict__ ctx_adj,
                        const int* __restrict__ resp_adj,
                        const float* __restrict__ ctx_h,
                        const float* __restrict__ resp_h,
                        const float* __restrict__ conv_b,
                        float* __restrict__ ctx_gcn,
                        float* __restrict__ resp_gcn) {
    __shared__ float sh[Sm * Cm];
    __shared__ float dinv[Sm];

    int g = blockIdx.x;
    const int* edges;
    const float* h;
    float* o;
    if (g < Bm * Tm) {
        edges = ctx_adj + (long)g * 2 * Em;
        h     = ctx_h   + (long)g * Sm * Cm;
        o     = ctx_gcn + (long)g * Sm * Cm;
    } else {
        long b = g - Bm * Tm;
        edges = resp_adj + b * 2 * Em;
        h     = resp_h   + b * Sm * Cm;
        o     = resp_gcn + b * Sm * Cm;
    }
    const int* src = edges;
    const int* dst = edges + Em;
    int tid = threadIdx.x;

    for (int i = tid; i < Sm * Cm; i += blockDim.x) sh[i] = h[i];
    if (tid < Sm) {
        float dg = 1.0f;                      // self-loop
        for (int e = 0; e < Em; ++e) dg += (dst[e] == tid) ? 1.0f : 0.0f;
        dinv[tid] = rsqrtf(fmaxf(dg, 1e-12f));
    }
    __syncthreads();

    if (tid < Sm) {
        float di = dinv[tid];
        float acc[Cm];
#pragma unroll
        for (int c = 0; c < Cm; ++c) acc[c] = sh[tid * Cm + c] * di * di;
        for (int e = 0; e < Em; ++e) {
            if (dst[e] == tid) {
                int s = src[e];
                float nm = dinv[s] * di;
#pragma unroll
                for (int c = 0; c < Cm; ++c) acc[c] += sh[s * Cm + c] * nm;
            }
        }
#pragma unroll
        for (int c = 0; c < Cm; ++c) o[tid * Cm + c] = acc[c] + conv_b[c];
    }
}

// ---------------------------------------------------------------------------
// K3: cross features: out[p,c] = sum_s att[p,s] * proj[s,c]  ([50,50]@[50,10])
// Blocks: 512 for u_cross (from resp proj), 2560 for r_cross (from ctx proj).
// ---------------------------------------------------------------------------
__global__ void gmn_cross(const float* __restrict__ att,
                          const float* __restrict__ resp_xp,
                          const float* __restrict__ ctx_xp,
                          float* __restrict__ u_cross,
                          float* __restrict__ r_cross) {
    __shared__ float proj[Sm * Cm];
    int g = blockIdx.x;
    const float* in;
    float* out;
    if (g < Bm) { in = resp_xp + (long)g * Sm * Cm; out = u_cross + (long)g * Sm * Cm; }
    else        { long k = g - Bm; in = ctx_xp + k * Sm * Cm; out = r_cross + k * Sm * Cm; }
    int tid = threadIdx.x;
    for (int i = tid; i < Sm * Cm; i += blockDim.x) proj[i] = in[i];
    __syncthreads();
    for (int i = tid; i < Sm * Cm; i += blockDim.x) {
        int p = i / Cm, c = i % Cm;
        float s = 0.f;
#pragma unroll
        for (int q = 0; q < Sm; ++q) s += att[p * Sm + q] * proj[q * Cm + c];
        out[i] = s;
    }
}

// ---------------------------------------------------------------------------
// K4: multi-perspective matching + feature fusion + FFN -> logits[b*T+t]
// One block per (b,t); threads 0..49 each own a node.
// ---------------------------------------------------------------------------
__global__ void gmn_mpm(const float* __restrict__ ctx_gcn,
                        const float* __restrict__ resp_gcn,
                        const float* __restrict__ u_cross,
                        const float* __restrict__ r_cross,
                        const float* __restrict__ assignW,
                        const float* __restrict__ W1, const float* __restrict__ b1,
                        const float* __restrict__ W2, const float* __restrict__ b2,
                        const float* __restrict__ fW1, const float* __restrict__ fb1,
                        const float* __restrict__ fW2, const float* __restrict__ fb2,
                        float* __restrict__ logits) {
    __shared__ float red[Sm * 20];
    __shared__ float gu[20], gr[20], feat[80], z[40];

    int g = blockIdx.x;           // b*T + t
    int b = g / Tm;
    int tid = threadIdx.x;

    for (int pass = 0; pass < 2; ++pass) {
        const float* selfp  = (pass == 0) ? ctx_gcn + (long)g * Sm * Cm
                                          : resp_gcn + (long)b * Sm * Cm;
        const float* crossp = (pass == 0) ? u_cross + (long)b * Sm * Cm
                                          : r_cross + (long)g * Sm * Cm;
        if (tid < Sm) {
            int s = tid;
            float sf[Cm];
            float na = 0.f, nc = 0.f, dotv = 0.f;
#pragma unroll
            for (int c = 0; c < Cm; ++c) {
                float as = assignW[s * Cm + c];
                sf[c] = selfp[s * Cm + c];
                float cv = crossp[s * Cm + c];
                float a = as * sf[c], cc = as * cv;
                dotv += a * cc; na += a * a; nc += cc * cc;
            }
            float cosv = dotv / fmaxf(sqrtf(na) * sqrtf(nc), 1e-8f);
            float h1[20];
#pragma unroll
            for (int j = 0; j < 20; ++j) {
                float v = b1[j] + cosv * W1[j];           // W1 row 0 = cos
#pragma unroll
                for (int c = 0; c < Cm; ++c) v += sf[c] * W1[(c + 1) * 20 + j];
                h1[j] = fmaxf(v, 0.f);
            }
#pragma unroll
            for (int j = 0; j < 20; ++j) {
                float v = b2[j];
#pragma unroll
                for (int k = 0; k < 20; ++k) v += h1[k] * W2[k * 20 + j];
                red[s * 20 + j] = v;
            }
        }
        __syncthreads();
        if (tid < 20) {
            float mx = red[tid];
            for (int s = 1; s < Sm; ++s) mx = fmaxf(mx, red[s * 20 + tid]);
            (pass == 0 ? gu : gr)[tid] = mx;
        }
        __syncthreads();
    }

    if (tid < 20) {
        float u = gu[tid], r = gr[tid];
        feat[tid]      = u;
        feat[20 + tid] = r;
        feat[40 + tid] = u * r;
        feat[60 + tid] = fabsf(u - r);
    }
    __syncthreads();
    if (tid < 40) {
        float v = fb1[tid];
#pragma unroll
        for (int k = 0; k < 80; ++k) v += feat[k] * fW1[k * 40 + tid];
        z[tid] = fmaxf(v, 0.f);
    }
    __syncthreads();
    if (tid == 0) {
        float v = fb2[0];
#pragma unroll
        for (int k = 0; k < 40; ++k) v += z[k] * fW2[k];
        logits[g] = v;
    }
}

// K5: sigmoid(sum over T)
__global__ void gmn_final(const float* __restrict__ logits, float* __restrict__ out) {
    int b = blockIdx.x * blockDim.x + threadIdx.x;
    if (b < Bm) {
        float s = 0.f;
#pragma unroll
        for (int t = 0; t < Tm; ++t) s += logits[b * Tm + t];
        out[b] = 1.0f / (1.0f + expf(-s));
    }
}

extern "C" void kernel_launch(void* const* d_in, const int* in_sizes, int n_in,
                              void* d_out, int out_size, void* d_ws, size_t ws_size,
                              hipStream_t stream) {
    const int*   ctx_in   = (const int*)d_in[0];
    const int*   resp_in  = (const int*)d_in[1];
    const int*   ctx_adj  = (const int*)d_in[2];
    const int*   resp_adj = (const int*)d_in[3];
    const float* emb      = (const float*)d_in[4];
    const float* convW    = (const float*)d_in[5];
    const float* convb    = (const float*)d_in[6];
    const float* crossW   = (const float*)d_in[7];
    const float* att      = (const float*)d_in[8];
    const float* assignW  = (const float*)d_in[9];
    const float* W1       = (const float*)d_in[10];
    const float* b1       = (const float*)d_in[11];
    const float* W2       = (const float*)d_in[12];
    const float* b2       = (const float*)d_in[13];
    const float* fW1      = (const float*)d_in[14];
    const float* fb1      = (const float*)d_in[15];
    const float* fW2      = (const float*)d_in[16];
    const float* fb2      = (const float*)d_in[17];

    char* ws = (char*)d_ws;
    size_t off = 0;
    auto alloc = [&](size_t bytes) -> void* {
        void* p = ws + off;
        off += (bytes + 255) & ~(size_t)255;
        return p;
    };

    const size_t CTX_ROWS  = (size_t)Bm * Tm * Sm;   // 128000
    const size_t RESP_ROWS = (size_t)Bm * Sm;        // 25600

    unsigned int* bfC      = (unsigned int*)alloc(BFRAG_DW * 4);
    unsigned int* bfX      = (unsigned int*)alloc(BFRAG_DW * 4);
    float* ctx_h    = (float*)alloc(CTX_ROWS  * Cm * 4);
    float* ctx_xp   = (float*)alloc(CTX_ROWS  * Cm * 4);
    float* resp_h   = (float*)alloc(RESP_ROWS * Cm * 4);
    float* resp_xp  = (float*)alloc(RESP_ROWS * Cm * 4);
    float* ctx_gcn  = (float*)alloc(CTX_ROWS  * Cm * 4);
    float* resp_gcn = (float*)alloc(RESP_ROWS * Cm * 4);
    float* u_cross  = (float*)alloc(RESP_ROWS * Cm * 4);
    float* r_cross  = (float*)alloc(CTX_ROWS  * Cm * 4);
    float* logits   = (float*)alloc((size_t)Bm * Tm * 4);

    gmn_prep_bfrag<<<24, 256, 0, stream>>>(convW, crossW, bfC, bfX);

    // 128000/16 = 8000 tiles, 8 waves/block -> 1000 blocks; resp: 1600 tiles
    gmn_proj<<<1000, 256, 0, stream>>>(ctx_in, 8000, emb, bfC, bfX, ctx_h, ctx_xp);
    gmn_proj<<<200,  256, 0, stream>>>(resp_in, 1600, emb, bfC, bfX, resp_h, resp_xp);

    gmn_gcn<<<Bm * Tm + Bm, 64, 0, stream>>>(ctx_adj, resp_adj, ctx_h, resp_h,
                                             convb, ctx_gcn, resp_gcn);

    gmn_cross<<<Bm + Bm * Tm, 128, 0, stream>>>(att, resp_xp, ctx_xp,
                                                u_cross, r_cross);

    gmn_mpm<<<Bm * Tm, 128, 0, stream>>>(ctx_gcn, resp_gcn, u_cross, r_cross,
                                         assignW, W1, b1, W2, b2,
                                         fW1, fb1, fW2, fb2, logits);

    gmn_final<<<(Bm + 255) / 256, 256, 0, stream>>>(logits, (float*)d_out);
}
